// AttentiveStatisticsPooling_16209206575321
// MI455X (gfx1250) — matrix operand order repaired and checked by reading, not compiled, not run on hardware
//
#include <hip/hip_runtime.h>
#include <hip/hip_bf16.h>

// CDNA5 / gfx1250 attentive-statistics pooling.
// B=64, T=2048, D=256. Bandwidth-bound (xs = 128MiB per pass, L2-resident);
// score GEMM (17.2 GFLOP) runs on v_wmma_f32_16x16x32_bf16.

typedef __attribute__((ext_vector_type(16))) __bf16 v16bf;
typedef __attribute__((ext_vector_type(8)))  float  v8f;

static constexpr int Bn = 64;
static constexpr int Tn = 2048;
static constexpr int Dn = 256;

// round-to-nearest-even fp32 -> bf16 (as ushort); avoids scalar __bf16 math
__device__ __forceinline__ unsigned int pk2bf(float a, float b) {
    unsigned int ua = __float_as_uint(a);
    unsigned int ub = __float_as_uint(b);
    ua = (ua + 0x7FFFu + ((ua >> 16) & 1u)) >> 16;
    ub = (ub + 0x7FFFu + ((ub >> 16) & 1u)) >> 16;
    return (ua & 0xFFFFu) | (ub << 16);
}

__device__ __forceinline__ float fast_tanh(float x) {
    // tanh(x) = 1 - 2/(exp(2x)+1); saturates correctly at +-1
    return 1.0f - 2.0f / (__expf(2.0f * x) + 1.0f);
}

// 32-bit LDS byte offset of a generic pointer to a __shared__ object
#define LDS_OFF(p) ((unsigned)(unsigned long long) \
        (__attribute__((address_space(3))) const void*)(p))

// ---------------------------------------------------------------------------
// K0: pre-swizzle W (fp32 [D][D], h = x @ W^T) into bf16 WMMA B-fragments.
// Fragment f = e_tile*8 + k covers B[k*32 .. k*32+31][e_tile*16 .. +15].
// B layout used here (32x16 bf16): lane n = col, lanes 0-15 hold K 0..15,
// lanes 16-31 hold K 16..31 -> 16 contiguous bf16 (32B) per lane.
// B[kk][n] = W[n][kk].
// ---------------------------------------------------------------------------
__global__ __launch_bounds__(256) void wprep_kernel(const float* __restrict__ W,
                                                    unsigned short* __restrict__ wfrag) {
    int g    = blockIdx.x * 256 + threadIdx.x;   // 0..4095 (128 frags * 32 lanes)
    int lane = g & 31;
    int frag = g >> 5;
    int et   = frag >> 3;
    int k    = frag & 7;
    int n    = et * 16 + (lane & 15);
    int kb   = k * 32 + ((lane & 16) ? 16 : 0);
    const float* src = W + n * Dn + kb;
    union { v16bf v; unsigned int u[8]; } w;
#pragma unroll
    for (int j = 0; j < 8; ++j)
        w.u[j] = pk2bf(src[2 * j], src[2 * j + 1]);
    *((v16bf*)(wfrag) + g) = w.v;                // 32B aligned store
}

// ---------------------------------------------------------------------------
// K1: scores[b][t] = attn . tanh(W x_t + bias).  512 blocks (64 b x 8 chunks),
// 256 threads = 8 waves.  Per 64-t tile: waves (tsub 0..3) x (e-half 0..1).
// ---------------------------------------------------------------------------
__global__ __launch_bounds__(256) void scores_kernel(const float* __restrict__ xs,
                                                     const float* __restrict__ bias,
                                                     const float* __restrict__ attn,
                                                     const unsigned short* __restrict__ wfrag,
                                                     float* __restrict__ scores_out) {
    __shared__ unsigned short xt[64 * 256];      // 32 KB: x tile as bf16
    __shared__ float sloc[256];                  // scores for this 256-t chunk

    const int tid     = threadIdx.x;
    const int b       = blockIdx.x >> 3;
    const int tchunk0 = (blockIdx.x & 7) * 256;

    const int lane  = tid & 31;
    const int wv    = tid >> 5;
    const int tsub  = wv & 3;                    // 16-t subtile within 64-t tile
    const int ehalf = wv >> 2;                   // 0 or 1 (e-tiles 0-7 / 8-15)
    const int eLane = lane & 15;
    const int hi    = lane >> 4;                 // half-wave select

    sloc[tid] = 0.0f;
    __syncthreads();

    for (int iter = 0; iter < 4; ++iter) {
        const int t0 = tchunk0 + iter * 64;
        const float* src = xs + ((size_t)b * Tn + t0) * Dn;

        // stage 64 x 256 fp32 -> bf16 into LDS, coalesced float4 loads
#pragma unroll
        for (int j = 0; j < 16; ++j) {
            int f = (tid + j * 256) * 4;         // flat element index, 0..16383
            float4 x4 = *(const float4*)(src + f);
            uint2 p;
            p.x = pk2bf(x4.x, x4.y);
            p.y = pk2bf(x4.z, x4.w);
            *(uint2*)(&xt[f]) = p;
        }
        if (iter + 1 < 4)
            __builtin_prefetch(src + 64 * 256, 0, 0);   // global_prefetch_b8
        __syncthreads();

        float sacc[8];
#pragma unroll
        for (int r = 0; r < 8; ++r) sacc[r] = 0.0f;

        const int tb = tsub * 16;
        // A layout (16x32 bf16): lane m=eLane; lanes<16 hold K 0..15,
        // lanes>=16 hold K 16..31 -> two 16B LDS chunks at +0 and +8.
        const unsigned short* abase = &xt[(tb + eLane) * 256 + hi * 16];
        const v16bf* bbase = (const v16bf*)wfrag;

        for (int et2 = 0; et2 < 8; ++et2) {
            const int etile = ehalf * 8 + et2;
            const int e     = etile * 16 + eLane;
            v8f c = {};
#pragma unroll
            for (int k = 0; k < 8; ++k) {
                union { v16bf v; uint4 q[2]; } af;
                const unsigned short* ap = abase + k * 32;
                af.q[0] = *(const uint4*)(ap);        // ds_load_b128
                af.q[1] = *(const uint4*)(ap + 8);    // ds_load_b128
                v16bf bm = bbase[(etile * 8 + k) * 32 + lane];
                c = __builtin_amdgcn_wmma_f32_16x16x32_bf16(
                        false, af.v, false, bm, (short)0, c, false, false);
            }
            const float be = bias[e];
            const float ae = attn[e];
#pragma unroll
            for (int r = 0; r < 8; ++r) {
                // C layout: lane<16 -> (t=tb+r, e), lane>=16 -> (t=tb+8+r, e)
                float h = c[r] + be;
                sacc[r] += ae * fast_tanh(h);
            }
        }

        // reduce the 16 e-lanes sharing each t (xor stays within half-wave)
#pragma unroll
        for (int r = 0; r < 8; ++r) {
            float v = sacc[r];
            v += __shfl_xor(v, 1, 32);
            v += __shfl_xor(v, 2, 32);
            v += __shfl_xor(v, 4, 32);
            v += __shfl_xor(v, 8, 32);
            if (eLane == 0)   // lane 0 -> t=tb+r ; lane 16 -> t=tb+8+r
                atomicAdd(&sloc[iter * 64 + tb + hi * 8 + r], v);  // ds_add_f32
        }
        __syncthreads();   // protects xt reuse and makes sloc visible
    }

    scores_out[(size_t)b * Tn + tchunk0 + tid] = sloc[tid];
}

// ---------------------------------------------------------------------------
// K2: masked softmax over T + weighted mean/std pooling.  64 blocks, 256 thr.
// Scores row (8KB) is pulled into LDS with the gfx1250 async-to-LDS pipe
// (ASYNCcnt); second xs pass is L2-resident (xs fits in 192MB L2).
// ---------------------------------------------------------------------------
__global__ __launch_bounds__(256) void pool_kernel(const float* __restrict__ xs,
                                                   const int* __restrict__ mask,
                                                   const float* __restrict__ scores_in,
                                                   float* __restrict__ out) {
    __shared__ float w_lds[Tn];                  // 8 KB
    __shared__ float red[256];

    const int tid = threadIdx.x;
    const int b   = blockIdx.x;

    // --- async copy scores row into LDS: 512 x 16B transfers, 2 per thread ---
    {
        const unsigned ldsbase = LDS_OFF(w_lds);
#pragma unroll
        for (int j = 0; j < 2; ++j) {
            const int i = tid + j * 256;                       // b128 chunk id
            unsigned long long g =
                (unsigned long long)(const void*)(scores_in + (size_t)b * Tn + i * 4);
            unsigned l = ldsbase + i * 16;
            asm volatile("global_load_async_to_lds_b128 %0, %1, off"
                         :: "v"(l), "v"(g) : "memory");
        }
        asm volatile("s_wait_asynccnt 0x0" ::: "memory");
    }
    __syncthreads();

    const float NEG = -3.0e38f;
    float lmax = NEG;
#pragma unroll
    for (int j = 0; j < 8; ++j) {
        int t = tid + j * 256;
        float s = w_lds[t];
        s = mask[(size_t)b * Tn + t] ? s : NEG;
        w_lds[t] = s;
        lmax = fmaxf(lmax, s);
    }
    red[tid] = lmax;
    __syncthreads();
    for (int s = 128; s > 0; s >>= 1) {
        if (tid < s) red[tid] = fmaxf(red[tid], red[tid + s]);
        __syncthreads();
    }
    const float m = red[0];
    __syncthreads();

    float lsum = 0.0f;
#pragma unroll
    for (int j = 0; j < 8; ++j) {
        int t = tid + j * 256;
        float e = __expf(w_lds[t] - m);          // masked entries -> exp(-huge)=0
        w_lds[t] = e;
        lsum += e;
    }
    red[tid] = lsum;
    __syncthreads();
    for (int s = 128; s > 0; s >>= 1) {
        if (tid < s) red[tid] += red[tid + s];
        __syncthreads();
    }
    const float inv = 1.0f / red[0];
    __syncthreads();
#pragma unroll
    for (int j = 0; j < 8; ++j) w_lds[tid + j * 256] *= inv;
    __syncthreads();

    // pooling: thread <-> feature d (perfectly coalesced per t row)
    const float* xb = xs + (size_t)b * Tn * Dn;
    float mu = 0.0f, s2 = 0.0f;
    for (int t = 0; t < Tn; ++t) {
        float wt = w_lds[t];
        float v  = xb[(size_t)t * Dn + tid];
        mu += wt * v;
        s2 += wt * v * v;
    }
    out[(size_t)b * 2 * Dn + tid]      = mu;
    out[(size_t)b * 2 * Dn + Dn + tid] = sqrtf(fmaxf(s2 - mu * mu, 1e-5f));
}

// ---------------------------------------------------------------------------
extern "C" void kernel_launch(void* const* d_in, const int* in_sizes, int n_in,
                              void* d_out, int out_size, void* d_ws, size_t ws_size,
                              hipStream_t stream) {
    (void)in_sizes; (void)n_in; (void)out_size; (void)ws_size;
    const float* xs   = (const float*)d_in[0];
    const int*   mask = (const int*)d_in[1];
    // d_in[2] (mask2) unused by the module
    const float* W    = (const float*)d_in[3];
    const float* bias = (const float*)d_in[4];
    const float* attn = (const float*)d_in[5];
    float* out = (float*)d_out;

    unsigned short* wfrag = (unsigned short*)d_ws;                 // 128 KB
    float* scores = (float*)((char*)d_ws + 128 * 1024);            // 512 KB

    wprep_kernel <<<16,  256, 0, stream>>>(W, wfrag);
    scores_kernel<<<512, 256, 0, stream>>>(xs, bias, attn, wfrag, scores);
    pool_kernel  <<<64,  256, 0, stream>>>(xs, mask, scores, out);
}
